// GraphModule_11879879541308
// MI455X (gfx1250) — matrix operand order repaired and checked by reading, not compiled, not run on hardware
//
#include <hip/hip_runtime.h>

// -------- problem constants (from reference) --------
constexpr int Bb = 8, Hh = 16, Dd = 128, NSPLIT = 32;
constexpr int ROWSTR  = Hh * Dd;   // 2048 halfs between consecutive keys
constexpr int RSTRIDE = 132;       // floats per partial record: acc[128], m, l, pad[2]
// SCALE applied to q and k => SCALE^2 on scores; fold log2(e) so exp -> v_exp_f32.
#define QSCALE (0.29730177875068026 * 0.29730177875068026 * 1.4426950408889634)

typedef __attribute__((ext_vector_type(16))) _Float16 v16h;
typedef __attribute__((ext_vector_type(8)))  _Float16 v8h;
typedef __attribute__((ext_vector_type(8)))  float    v8f;
typedef __attribute__((ext_vector_type(4)))  float    v4f;

union vh16u { v16h v; v8h h[2]; };

// ---------------------------------------------------------------------------
// Kernel 1: one wave per (b, h, split). Streams its KV slice exactly once.
//   QK^T : v_wmma_f32_16x16x32_f16, A = 16-key tile (A-frag loaded straight
//          from global via B128 loads), B = q replicated into all 16 columns.
//   PV   : VALU fma on the SAME registers (K == V tensor), fp32 accum.
// Partial (acc[128], m, l) written to workspace for the split-K merge.
// ---------------------------------------------------------------------------
__global__ __launch_bounds__(256) void attn_partial(
    const _Float16* __restrict__ q,
    const _Float16* __restrict__ kv,
    const int*      __restrict__ cpos,
    float*          __restrict__ part,
    int maxlen)
{
  const int lane  = threadIdx.x & 31;
  const int wave  = (int)((blockIdx.x * blockDim.x + threadIdx.x) >> 5);
  const int split = wave & (NSPLIT - 1);
  const int bh    = wave / NSPLIT;          // b*H + h
  const int b     = bh / Hh;
  const int h     = bh % Hh;
  const int hi    = (lane >> 4) & 1;        // which 16-lane half
  const int r15   = lane & 15;

  const int L   = cpos[0] + 1;
  const int Lm1 = L - 1;
  int kps  = ((L + NSPLIT - 1) / NSPLIT + 15) & ~15;  // keys per split, 16-aligned
  int k0   = split * kps;
  int kend = k0 + kps; if (kend > L) kend = L;
  int nt   = (kend > k0) ? ((kend - k0 + 15) >> 4) : 0;

  // ---- q as WMMA B-frag: lanes 0-15 hold K={0..7,16..23}, lanes 16-31 K={8..15,24..31} of
  //      each 32-dim chunk; all 16 columns identical (q broadcast). Pre-scaled.
  vh16u qf[4];
  {
    const _Float16* qp = q + (long)bh * Dd;
#pragma unroll
    for (int c = 0; c < 4; ++c) {
      vh16u raw;
      raw.h[0] = *(const v8h*)(qp + c * 32 + hi * 8);       // B128
      raw.h[1] = *(const v8h*)(qp + c * 32 + 16 + hi * 8);  // B128
#pragma unroll
      for (int e = 0; e < 16; ++e)
        qf[c].v[e] = (_Float16)((float)raw.v[e] * (float)QSCALE);
    }
  }

  const _Float16* kb = kv + ((long)b * maxlen) * ROWSTR + (long)h * Dd;

  vh16u ac[4], an[4];
  // A-frag of a 16-key tile: lane l (l<16) holds key row l, dims {32c+0..7,
  // 32c+16..23}; lane l+16 holds row l, dims {32c+8..15, 32c+24..31}.
  auto loadTile = [&](int tk, vh16u* a) {
    int rk = tk + r15; if (rk > Lm1) rk = Lm1;            // clamp tail rows
    const _Float16* rp = kb + (long)rk * ROWSTR;
#pragma unroll
    for (int c = 0; c < 4; ++c) {
      a[c].h[0] = *(const v8h*)(rp + c * 32 + hi * 8);        // B128
      a[c].h[1] = *(const v8h*)(rp + c * 32 + 16 + hi * 8);   // B128
    }
  };

  float mrun = -__builtin_inff();
  float lrun = 0.0f;
  float acc[64];
#pragma unroll
  for (int i = 0; i < 64; ++i) acc[i] = 0.0f;

  if (nt > 0) loadTile(k0, ac);

  for (int t = 0; t < nt; ++t) {
    const int tk = k0 + (t << 4);
    if (t + 1 < nt) loadTile(tk + 16, an);   // pipeline next tile's loads

    // L2-warm tile t+2 at zero VGPR cost (row = two 128B lines).
    if (t + 2 < nt) {
      int rk2 = tk + 32 + r15; if (rk2 > Lm1) rk2 = Lm1;
      const _Float16* rp2 = kb + (long)rk2 * ROWSTR;
      __builtin_prefetch(rp2, 0, 0);
      __builtin_prefetch(rp2 + 64, 0, 0);
    }

    // scores: C[m,n] identical across n. lane<16: reg r = key tk+r;
    // lane>=16: reg r = key tk+8+r. (log2 domain)
    v8f cs = {};
#pragma unroll
    for (int c = 0; c < 4; ++c)
      cs = __builtin_amdgcn_wmma_f32_16x16x32_f16(false, ac[c].v, false, qf[c].v,
                                                  (short)0, cs, false, false);

    if (tk + 16 > L) {                        // tail mask (uniform branch)
#pragma unroll
      for (int r = 0; r < 8; ++r)
        if (tk + (hi << 3) + r >= L) cs[r] = -__builtin_inff();
    }

    // ---- online softmax (wave-uniform m, l) ----
    float tmax = cs[0];
#pragma unroll
    for (int r = 1; r < 8; ++r) tmax = fmaxf(tmax, cs[r]);
    tmax = fmaxf(tmax, __shfl_xor(tmax, 16, 32));
    const float mnew = fmaxf(mrun, tmax);
    const float corr = exp2f(mrun - mnew);

    float e[8];
    float tl = 0.0f;
#pragma unroll
    for (int r = 0; r < 8; ++r) { e[r] = exp2f(cs[r] - mnew); tl += e[r]; }
    tl += __shfl_xor(tl, 16, 32);
    lrun = lrun * corr + tl;
    mrun = mnew;

    // ---- pick p for this lane's owned key (row l&15) ----
    // own regs cover keys {0..7} (lanes<16) or {8..15} (lanes>=16);
    // select e[(l&15)&7] via cndmask tree, then one xor-16 shuffle if the
    // wanted key lives in the other half.
    float a0_ = (r15 & 1) ? e[1] : e[0];
    float a1_ = (r15 & 1) ? e[3] : e[2];
    float a2_ = (r15 & 1) ? e[5] : e[4];
    float a3_ = (r15 & 1) ? e[7] : e[6];
    float b0_ = (r15 & 2) ? a1_ : a0_;
    float b1_ = (r15 & 2) ? a3_ : a2_;
    float sel = (r15 & 4) ? b1_ : b0_;
    float cross = __shfl_xor(sel, 16, 32);
    float myp = ((((r15 >> 3) ^ hi) & 1) ? cross : sel);

    // ---- rescale only when the running max actually moved (uniform) ----
    if (corr < 1.0f) {
#pragma unroll
      for (int i = 0; i < 64; ++i) acc[i] *= corr;
    }
    // ---- PV: V rows are already in the A-frag registers (K == V) ----
#pragma unroll
    for (int c = 0; c < 4; ++c)
#pragma unroll
      for (int e2 = 0; e2 < 16; ++e2)
        acc[c * 16 + e2] = fmaf(myp, (float)ac[c].v[e2], acc[c * 16 + e2]);

#pragma unroll
    for (int c = 0; c < 4; ++c) ac[c].v = an[c].v;
  }

  // ---- butterfly-reduce the 16 key-rows (lanes) per half-wave ----
  // afterwards every lane in a half holds the reduced values for that half.
#pragma unroll
  for (int i = 0; i < 64; ++i) {
    acc[i] += __shfl_xor(acc[i], 1, 32);
    acc[i] += __shfl_xor(acc[i], 2, 32);
    acc[i] += __shfl_xor(acc[i], 4, 32);
    acc[i] += __shfl_xor(acc[i], 8, 32);
  }

  // ---- coalesced partial store: each lane stores one aligned float4 ----
  // flat acc index f = r15*4 -> chunk c = f>>4, e2 base = f&15 (multiple of 4);
  // dim = 32c + 8*hi + e2b (+8 if e2b >= 8); record is 16B aligned (132 floats).
  float* pp = part + ((long)bh * NSPLIT + split) * RSTRIDE;
  {
    const int f   = r15 * 4;
    const int c   = f >> 4;
    const int e2b = f & 15;
    const int d   = c * 32 + hi * 8 + e2b + ((e2b >= 8) ? 8 : 0);
    v4f o = { acc[f], acc[f + 1], acc[f + 2], acc[f + 3] };
    *(v4f*)(pp + d) = o;
  }
  if (lane == 0) { pp[Dd] = mrun; pp[Dd + 1] = lrun; }
}

// ---------------------------------------------------------------------------
// Kernel 2: split-K merge. 128 blocks (b,h) x 128 threads (d).
// ---------------------------------------------------------------------------
__global__ __launch_bounds__(128) void attn_reduce(
    const float* __restrict__ part, float* __restrict__ out)
{
  const int bh = blockIdx.x;
  const int d  = threadIdx.x;
  const float* p0 = part + (long)bh * NSPLIT * RSTRIDE;

  float M = -__builtin_inff();
#pragma unroll
  for (int i = 0; i < NSPLIT; ++i) M = fmaxf(M, p0[i * RSTRIDE + Dd]);

  float Ls = 0.0f, o = 0.0f;
#pragma unroll
  for (int i = 0; i < NSPLIT; ++i) {
    float w = exp2f(p0[i * RSTRIDE + Dd] - M);     // log2 domain throughout
    Ls += p0[i * RSTRIDE + Dd + 1] * w;
    o  += w * p0[i * RSTRIDE + d];
  }
  out[(long)bh * Dd + d] = o / Ls;
}

// ---------------------------------------------------------------------------
extern "C" void kernel_launch(void* const* d_in, const int* in_sizes, int n_in,
                              void* d_out, int out_size, void* d_ws, size_t ws_size,
                              hipStream_t stream)
{
  (void)n_in; (void)out_size; (void)ws_size;
  const _Float16* q  = (const _Float16*)d_in[0];
  const _Float16* kv = (const _Float16*)d_in[1];
  const int* cpos    = (const int*)d_in[2];
  float* out  = (float*)d_out;
  float* part = (float*)d_ws;   // 4096 * 132 floats ~ 2.2 MB

  const int maxlen = in_sizes[1] / (Bb * Hh * Dd);   // 32768
  const int waves  = Bb * Hh * NSPLIT;               // 4096
  const int block  = 256;                            // 8 waves / workgroup
  const int grid   = (waves * 32) / block;           // 512

  attn_partial<<<grid, block, 0, stream>>>(q, kv, cpos, part, maxlen);
  attn_reduce<<<Bb * Hh, Dd, 0, stream>>>(part, out);
}
